// GNN_Block_38027640439448
// MI455X (gfx1250) — matrix operand order repaired and checked by reading, not compiled, not run on hardware
//
#include <hip/hip_runtime.h>
#include <hip/hip_fp16.h>

typedef __attribute__((ext_vector_type(16))) _Float16 v16h;
typedef __attribute__((ext_vector_type(8)))  float    v8f;

#define NEG_SLOPE 0.2f
#define BN_EPS    1e-5f
#define SM_EPS    1e-16f

// ---- order-preserving float<->uint encoding for atomicMax-based segment_max
__device__ __forceinline__ unsigned fenc(float f) {
    unsigned u = __float_as_uint(f);
    return (u & 0x80000000u) ? ~u : (u | 0x80000000u);
}
__device__ __forceinline__ float fdec(unsigned u) {
    return __uint_as_float((u & 0x80000000u) ? (u & 0x7fffffffu) : ~u);
}

// =====================================================================
// WMMA GEMM: C[M x Nout] = A[M x K] @ W[K x Nout] + b   (fp32 in, f16 MAC,
// f32 accumulate). One wave computes a 16x64 output strip = 4 adjacent
// 16x16 tiles, reusing the A fragment across 4 WMMAs per k-step.
// Requires: M % 16 == 0, K % 32 == 0, Nout % 64 == 0 (true for all launches).
// =====================================================================
__global__ void gemm_bias_wmma(const float* __restrict__ A,
                               const float* __restrict__ W,
                               const float* __restrict__ bias,
                               float* __restrict__ C,
                               int M, int K, int Nout)
{
    const int wave   = (blockIdx.x * blockDim.x + threadIdx.x) >> 5;
    const int lane   = threadIdx.x & 31;
    const int strips = Nout >> 6;           // 64-wide output strips
    const int rowTile = wave / strips;      // wave-uniform
    const int strip   = wave % strips;
    if (rowTile * 16 >= M) return;          // uniform exit: EXEC stays all-1s

    const int half  = lane >> 4;
    const int l15   = lane & 15;
    const int mrow  = rowTile * 16 + l15;
    const int kbA   = half ? 16 : 0;        // A-frag K base (contiguous, matches B)
    const int kbB   = half ? 16 : 0;        // B-frag K base (contiguous, matches A)
    const int ncol0 = strip * 64 + l15;

    v8f acc0 = {}, acc1 = {}, acc2 = {}, acc3 = {};
    for (int kt = 0; kt < K; kt += 32) {
        const float* Arow = A + (size_t)mrow * K + kt;
        v16h a;
#pragma unroll
        for (int j = 0; j < 16; ++j) {
            // element j of A frag holds K = kbA + j
            int ka = kbA + j;
            a[j] = (_Float16)Arow[ka];
        }
        const float* Wk = W + (size_t)(kt + kbB) * Nout + ncol0;
        v16h b0, b1, b2, b3;
#pragma unroll
        for (int j = 0; j < 16; ++j) {      // element j of B frag holds K = kbB + j
            const float* wr = Wk + (size_t)j * Nout;
            b0[j] = (_Float16)wr[0];
            b1[j] = (_Float16)wr[16];
            b2[j] = (_Float16)wr[32];
            b3[j] = (_Float16)wr[48];
        }
        acc0 = __builtin_amdgcn_wmma_f32_16x16x32_f16(false, a, false, b0, (short)0, acc0, false, false);
        acc1 = __builtin_amdgcn_wmma_f32_16x16x32_f16(false, a, false, b1, (short)0, acc1, false, false);
        acc2 = __builtin_amdgcn_wmma_f32_16x16x32_f16(false, a, false, b2, (short)0, acc2, false, false);
        acc3 = __builtin_amdgcn_wmma_f32_16x16x32_f16(false, a, false, b3, (short)0, acc3, false, false);
    }

    const float bv0 = bias[ncol0];
    const float bv1 = bias[ncol0 + 16];
    const float bv2 = bias[ncol0 + 32];
    const float bv3 = bias[ncol0 + 48];
    const int rbase = rowTile * 16 + (half ? 8 : 0);
#pragma unroll
    for (int r = 0; r < 8; ++r) {           // D layout: VGPR r -> M = rbase + r
        float* crow = C + (size_t)(rbase + r) * Nout + ncol0;
        crow[0]  = acc0[r] + bv0;
        crow[16] = acc1[r] + bv1;
        crow[32] = acc2[r] + bv2;
        crow[48] = acc3[r] + bv3;
    }
}

// =====================================================================
// Edge pass 1: logits[e,h] = sum_c att[h,c]*leakyrelu(xl[src,h,c]+xr[dst,h,c])
//              segment_max into mEnc[dst,h] (encoded uint atomicMax)
// =====================================================================
__global__ void edge_logits_max(const int* __restrict__ src, const int* __restrict__ dst,
                                int E, int Etot, int H, int C,
                                const float* __restrict__ xl, const float* __restrict__ xr,
                                const float* __restrict__ att,
                                float* __restrict__ logits, unsigned* __restrict__ mEnc)
{
    int idx = blockIdx.x * blockDim.x + threadIdx.x;
    if (idx >= Etot * H) return;
    int e = idx / H, h = idx - e * H;
    int s = (e < E) ? src[e] : (e - E);     // appended self-loops
    int d = (e < E) ? dst[e] : (e - E);
    const float* pl = xl + ((size_t)s * H + h) * C;
    const float* pr = xr + ((size_t)d * H + h) * C;
    const float* pa = att + (size_t)h * C;
    float acc = 0.f;
#pragma unroll 8
    for (int c = 0; c < C; ++c) {
        float v = pl[c] + pr[c];
        v = (v >= 0.f) ? v : NEG_SLOPE * v;
        acc += v * pa[c];
    }
    logits[idx] = acc;
    atomicMax(&mEnc[(size_t)d * H + h], fenc(acc));
}

// =====================================================================
// Edge pass 2: expv = exp(logit - m[dst]); segment_sum into denom[dst]
// =====================================================================
__global__ void edge_expsum(const int* __restrict__ dst, int E, int Etot, int H,
                            const float* __restrict__ logits,
                            const unsigned* __restrict__ mEnc,
                            float* __restrict__ expv, float* __restrict__ denom)
{
    int idx = blockIdx.x * blockDim.x + threadIdx.x;
    if (idx >= Etot * H) return;
    int e = idx / H, h = idx - e * H;
    int d = (e < E) ? dst[e] : (e - E);
    float m  = fdec(mEnc[(size_t)d * H + h]);
    float ev = __expf(logits[idx] - m);
    expv[idx] = ev;
    atomicAdd(&denom[(size_t)d * H + h], ev);
}

// =====================================================================
// Edge pass 3: alpha = expv/(denom[dst]+eps); accum[dst,h,:] += xl[src,h,:]*alpha
// =====================================================================
__global__ void edge_aggregate(const int* __restrict__ src, const int* __restrict__ dst,
                               int E, int Etot, int H, int C,
                               const float* __restrict__ xl,
                               const float* __restrict__ expv,
                               const float* __restrict__ denom,
                               float* __restrict__ accum)
{
    int idx = blockIdx.x * blockDim.x + threadIdx.x;
    if (idx >= Etot * H) return;
    int e = idx / H, h = idx - e * H;
    int s = (e < E) ? src[e] : (e - E);
    int d = (e < E) ? dst[e] : (e - E);
    float alpha = expv[idx] / (denom[(size_t)d * H + h] + SM_EPS);
    const float* pl = xl    + ((size_t)s * H + h) * C;
    float*       po = accum + ((size_t)d * H + h) * C;
#pragma unroll 4
    for (int c = 0; c < C; ++c)
        atomicAdd(&po[c], pl[c] * alpha);
}

// =====================================================================
// Mean over heads (H=1 => identity) + per-channel bias
// =====================================================================
__global__ void head_mean_bias(const float* __restrict__ accum,
                               const float* __restrict__ bias,
                               float* __restrict__ out, int N, int H, int C)
{
    int idx = blockIdx.x * blockDim.x + threadIdx.x;
    if (idx >= N * C) return;
    int n = idx / C, c = idx - n * C;
    float s = 0.f;
    for (int h = 0; h < H; ++h)
        s += accum[((size_t)n * H + h) * C + c];
    out[idx] = s / (float)H + bias[c];
}

// =====================================================================
// BatchNorm stats: per-channel sum / sumsq (blockDim.x == C; coalesced rows)
// =====================================================================
__global__ void bn_stats(const float* __restrict__ x, int N, int C,
                         float* __restrict__ sum, float* __restrict__ sumsq)
{
    int c = threadIdx.x;
    float s = 0.f, q = 0.f;
    for (int n = blockIdx.x; n < N; n += gridDim.x) {
        float v = x[(size_t)n * C + c];
        s += v; q += v * v;
    }
    atomicAdd(&sum[c], s);
    atomicAdd(&sumsq[c], q);
}

// =====================================================================
// y = relu(gamma*(x-mean)*rsqrt(var+eps)+beta), biased var (torch-style)
// =====================================================================
__global__ void bn_relu(const float* __restrict__ x,
                        const float* __restrict__ sum, const float* __restrict__ sumsq,
                        const float* __restrict__ g, const float* __restrict__ beta,
                        float* __restrict__ y, int N, int C)
{
    int idx = blockIdx.x * blockDim.x + threadIdx.x;
    if (idx >= N * C) return;
    int c = idx % C;
    float mean = sum[c] / (float)N;
    float var  = sumsq[c] / (float)N - mean * mean;
    float v = g[c] * (x[idx] - mean) * rsqrtf(var + BN_EPS) + beta[c];
    y[idx] = (v > 0.f) ? v : 0.f;
}

// =====================================================================
static inline int cdiv(long long a, long long b) { return (int)((a + b - 1) / b); }

extern "C" void kernel_launch(void* const* d_in, const int* in_sizes, int n_in,
                              void* d_out, int out_size, void* d_ws, size_t ws_size,
                              hipStream_t stream) {
    const float* x    = (const float*)d_in[0];
    const int*   ei   = (const int*)d_in[1];
    const float* Wl1  = (const float*)d_in[2];
    const float* bl1  = (const float*)d_in[3];
    const float* Wr1  = (const float*)d_in[4];
    const float* br1  = (const float*)d_in[5];
    const float* att1 = (const float*)d_in[6];
    const float* bias1= (const float*)d_in[7];
    const float* g1   = (const float*)d_in[8];
    const float* beta1= (const float*)d_in[9];
    const float* Wl2  = (const float*)d_in[10];
    const float* bl2  = (const float*)d_in[11];
    const float* Wr2  = (const float*)d_in[12];
    const float* br2  = (const float*)d_in[13];
    const float* att2 = (const float*)d_in[14];
    const float* bias2= (const float*)d_in[15];
    const float* g2   = (const float*)d_in[16];
    const float* beta2= (const float*)d_in[17];

    const int Fin = 128, Hid = 64, Out = 64, H1 = 4;
    const int N   = in_sizes[0] / Fin;      // 50000, divisible by 16
    const int E   = in_sizes[1] / 2;
    const int Et  = E + N;                  // with self-loops
    const int* srcE = ei;
    const int* dstE = ei + E;

    // ---- carve workspace -------------------------------------------------
    char* w = (char*)d_ws;
    auto carve = [&](size_t bytes) -> void* {
        void* p = (void*)w;
        w += (bytes + 255) & ~(size_t)255;
        return p;
    };
    float*    xl1    = (float*)   carve((size_t)N * H1 * Hid * 4);
    float*    xr1    = (float*)   carve((size_t)N * H1 * Hid * 4);
    float*    lg1    = (float*)   carve((size_t)Et * H1 * 4);
    float*    ev1    = (float*)   carve((size_t)Et * H1 * 4);
    unsigned* mE1    = (unsigned*)carve((size_t)N * H1 * 4);
    float*    dn1    = (float*)   carve((size_t)N * H1 * 4);
    float*    ac1    = (float*)   carve((size_t)N * H1 * Hid * 4);
    float*    h1pre  = (float*)   carve((size_t)N * Hid * 4);
    float*    h1     = (float*)   carve((size_t)N * Hid * 4);
    float*    xl2    = (float*)   carve((size_t)N * Out * 4);
    float*    xr2    = (float*)   carve((size_t)N * Out * 4);
    float*    lg2    = (float*)   carve((size_t)Et * 4);
    float*    ev2    = (float*)   carve((size_t)Et * 4);
    unsigned* mE2    = (unsigned*)carve((size_t)N * 4);
    float*    dn2    = (float*)   carve((size_t)N * 4);
    float*    ac2    = (float*)   carve((size_t)N * Out * 4);
    float*    o2pre  = (float*)   carve((size_t)N * Out * 4);
    float*    st1    = (float*)   carve(2 * Hid * 4);   // sum | sumsq
    float*    st2    = (float*)   carve(2 * Out * 4);
    (void)ws_size; (void)n_in;

    // ---- zero-init accumulators (capture-safe async memsets) -------------
    hipMemsetAsync(mE1, 0, (size_t)N * H1 * 4, stream);
    hipMemsetAsync(dn1, 0, (size_t)N * H1 * 4, stream);
    hipMemsetAsync(ac1, 0, (size_t)N * H1 * Hid * 4, stream);
    hipMemsetAsync(mE2, 0, (size_t)N * 4, stream);
    hipMemsetAsync(dn2, 0, (size_t)N * 4, stream);
    hipMemsetAsync(ac2, 0, (size_t)N * Out * 4, stream);
    hipMemsetAsync(st1, 0, 2 * Hid * 4, stream);
    hipMemsetAsync(st2, 0, 2 * Out * 4, stream);

    const int BLK = 256;

    // =========================== layer 1 =================================
    {   // xl1 = x@Wl1+bl1 ; xr1 = x@Wr1+br1   [N x 256]
        int K = Fin, No = H1 * Hid;
        long long waves = (long long)cdiv(N, 16) * (No / 64);
        int grid = cdiv(waves * 32, BLK);
        gemm_bias_wmma<<<grid, BLK, 0, stream>>>(x, Wl1, bl1, xl1, N, K, No);
        gemm_bias_wmma<<<grid, BLK, 0, stream>>>(x, Wr1, br1, xr1, N, K, No);
    }
    {
        long long nt = (long long)Et * H1;
        int grid = cdiv(nt, BLK);
        edge_logits_max<<<grid, BLK, 0, stream>>>(srcE, dstE, E, Et, H1, Hid,
                                                  xl1, xr1, att1, lg1, mE1);
        edge_expsum<<<grid, BLK, 0, stream>>>(dstE, E, Et, H1, lg1, mE1, ev1, dn1);
        edge_aggregate<<<grid, BLK, 0, stream>>>(srcE, dstE, E, Et, H1, Hid,
                                                 xl1, ev1, dn1, ac1);
    }
    head_mean_bias<<<cdiv((long long)N * Hid, BLK), BLK, 0, stream>>>(
        ac1, bias1, h1pre, N, H1, Hid);
    bn_stats<<<256, Hid, 0, stream>>>(h1pre, N, Hid, st1, st1 + Hid);
    bn_relu<<<cdiv((long long)N * Hid, BLK), BLK, 0, stream>>>(
        h1pre, st1, st1 + Hid, g1, beta1, h1, N, Hid);

    // =========================== layer 2 =================================
    {   // xl2/xr2 = h1@Wl2+bl2 / h1@Wr2+br2   [N x 64]
        int K = Hid, No = Out;
        long long waves = (long long)cdiv(N, 16) * (No / 64);
        int grid = cdiv(waves * 32, BLK);
        gemm_bias_wmma<<<grid, BLK, 0, stream>>>(h1, Wl2, bl2, xl2, N, K, No);
        gemm_bias_wmma<<<grid, BLK, 0, stream>>>(h1, Wr2, br2, xr2, N, K, No);
    }
    {
        int grid = cdiv((long long)Et, BLK);
        edge_logits_max<<<grid, BLK, 0, stream>>>(srcE, dstE, E, Et, 1, Out,
                                                  xl2, xr2, att2, lg2, mE2);
        edge_expsum<<<grid, BLK, 0, stream>>>(dstE, E, Et, 1, lg2, mE2, ev2, dn2);
        edge_aggregate<<<grid, BLK, 0, stream>>>(srcE, dstE, E, Et, 1, Out,
                                                 xl2, ev2, dn2, ac2);
    }
    head_mean_bias<<<cdiv((long long)N * Out, BLK), BLK, 0, stream>>>(
        ac2, bias2, o2pre, N, 1, Out);
    bn_stats<<<256, Out, 0, stream>>>(o2pre, N, Out, st2, st2 + Out);
    bn_relu<<<cdiv((long long)N * Out, BLK), BLK, 0, stream>>>(
        o2pre, st2, st2 + Out, g2, beta2, (float*)d_out, N, Out);
}
